// RegularizedAttention_4415226380583
// MI455X (gfx1250) — matrix operand order repaired and checked by reading, not compiled, not run on hardware
//
#include <hip/hip_runtime.h>
#include <hip/hip_bf16.h>

typedef __attribute__((ext_vector_type(16))) _Float16 v16h;
typedef __attribute__((ext_vector_type(4)))  _Float16 v4h;
typedef __attribute__((ext_vector_type(8)))  float    v8f;

#define HIDDEN 1024
#define SEQ    2048
#define BATCH  2
#define HEADS  16
#define HD     64
#define MROWS  (BATCH * SEQ)   // 4096
#define LDP    72              // padded LDS row stride (halves); 144B = 9 x 16B

static __device__ __forceinline__ v8f wmma16(v16h a, v16h b, v8f c) {
  // D = A(16x32,f16) * B(32x16,f16) + C(16x16,f32)
  return __builtin_amdgcn_wmma_f32_16x16x32_f16(false, a, false, b, (short)0, c, false, false);
}

// ---- fragment loaders (contiguous k order: k = 16*(lane/16) + i, same for A and B) ----
// A 16x32 (MxK): lanes 0-15 -> row M=lane, K=0..15; lanes 16-31 -> K=16..31
static __device__ __forceinline__ v16h load_a_f16g(const _Float16* __restrict__ src, int ld, int m0, int kb, int lane) {
  const _Float16* p = src + (size_t)(m0 + (lane & 15)) * ld + kb + ((lane >> 4) << 4);
  v16h a;
#pragma unroll
  for (int i = 0; i < 16; ++i) a[i] = p[i];
  return a;
}
static __device__ __forceinline__ v16h load_a_lds(const _Float16* t, int ld, int kb, int lane) {
  const _Float16* p = t + (lane & 15) * ld + kb + ((lane >> 4) << 4);
  v16h a;
#pragma unroll
  for (int i = 0; i < 16; ++i) a[i] = p[i];
  return a;
}
// B 32x16 (KxN) with B[k][n] = W[n][k] (W row-major [N,ld] f16); lane = N col, K group by lane>>4
static __device__ __forceinline__ v16h load_bT_f16(const _Float16* __restrict__ W, int ld, int n0, int kb, int lane) {
  const _Float16* p = W + (size_t)(n0 + (lane & 15)) * ld + kb + ((lane >> 4) << 4);
  v16h b;
#pragma unroll
  for (int i = 0; i < 16; ++i) b[i] = p[i];
  return b;
}
// scores B: B[d][kcol] = Ktile[kcol][d]  (LDS row-major [k][d]) -> contiguous reads
static __device__ __forceinline__ v16h load_b_kt(const _Float16* t, int ncol0, int db, int lane) {
  const _Float16* p = t + (ncol0 + (lane & 15)) * LDP + db + ((lane >> 4) << 4);
  v16h b;
#pragma unroll
  for (int i = 0; i < 16; ++i) b[i] = p[i];
  return b;
}
// PV B: B[k][d] = VtT[d][k] (LDS transposed [d][k]) -> contiguous reads
static __device__ __forceinline__ v16h load_b_vtT(const _Float16* t, int kb, int d0, int lane) {
  const _Float16* p = t + (size_t)(d0 + (lane & 15)) * LDP + kb + ((lane >> 4) << 4);
  v16h b;
#pragma unroll
  for (int i = 0; i < 16; ++i) b[i] = p[i];
  return b;
}

// =============== Kernel 0: bulk f32 -> f16 conversion (vectorized) ===============
__global__ void __launch_bounds__(256) cvt_f16_kernel(const float4* __restrict__ src,
                                                      v4h* __restrict__ dst, int n4) {
  int i = blockIdx.x * 256 + threadIdx.x;
  if (i < n4) {
    float4 v = src[i];
    v4h o;
    o[0] = (_Float16)v.x; o[1] = (_Float16)v.y; o[2] = (_Float16)v.z; o[3] = (_Float16)v.w;
    dst[i] = o;
  }
}

// =============== Kernel 1: fused QKV projection (all-f16 operands, f16 head-layout out) ===============
// Each wave: 32x64 output tile (2 A frags x 4 B frags -> 8 WMMAs per k-step)
__global__ void __launch_bounds__(128) qkv_proj_kernel(
    const _Float16* __restrict__ Xq, const _Float16* __restrict__ Xk, const _Float16* __restrict__ Xv,
    const _Float16* __restrict__ Wqh, const _Float16* __restrict__ Wkh, const _Float16* __restrict__ Wvh,
    const float* __restrict__ bq, const float* __restrict__ bk, const float* __restrict__ bv,
    _Float16* __restrict__ Qh, _Float16* __restrict__ Kh, _Float16* __restrict__ Vh) {
  int id = blockIdx.x;
  int nt = id & 15; id >>= 4;
  int mt = id & 31; id >>= 5;
  int which = id;                       // 0=Q 1=K 2=V
  const _Float16* X = (which == 0) ? Xq : (which == 1) ? Xk : Xv;
  const _Float16* W = (which == 0) ? Wqh : (which == 1) ? Wkh : Wvh;
  const float* bias = (which == 0) ? bq : (which == 1) ? bk : bv;
  _Float16* dst = (which == 0) ? Qh : (which == 1) ? Kh : Vh;
  const float oscale = (which == 0) ? 0.125f : 1.0f;   // fold sm_scale = 1/sqrt(64) into Q

  int lane = threadIdx.x & 31, wv = threadIdx.x >> 5;
  int m0 = mt * 128 + wv * 32;          // 32 rows per wave
  int n0 = nt * 64;
  v8f acc[8] = {{0},{0},{0},{0},{0},{0},{0},{0}};
#pragma unroll 4
  for (int kb = 0; kb < HIDDEN; kb += 32) {
    v16h a0 = load_a_f16g(X, HIDDEN, m0, kb, lane);
    v16h a1 = load_a_f16g(X, HIDDEN, m0 + 16, kb, lane);
    v16h bfr[4];
#pragma unroll
    for (int nf = 0; nf < 4; ++nf) bfr[nf] = load_bT_f16(W, HIDDEN, n0 + nf * 16, kb, lane);
#pragma unroll
    for (int nf = 0; nf < 4; ++nf) acc[nf] = wmma16(a0, bfr[nf], acc[nf]);
#pragma unroll
    for (int nf = 0; nf < 4; ++nf) acc[4 + nf] = wmma16(a1, bfr[nf], acc[4 + nf]);
  }
  // epilogue: bias, scale, scatter to [B,H,S,Hd] f16
  int rbase = (lane >> 4) << 3;
  int cl = lane & 15;
#pragma unroll
  for (int g = 0; g < 2; ++g) {
#pragma unroll
    for (int nf = 0; nf < 4; ++nf) {
#pragma unroll
      for (int r = 0; r < 8; ++r) {
        int m = m0 + g * 16 + rbase + r;        // 0..4095
        int n = n0 + nf * 16 + cl;              // 0..1023
        float y = (acc[g * 4 + nf][r] + bias[n]) * oscale;
        int b_ = m >> 11, s = m & (SEQ - 1);
        int h = n >> 6,  d = n & (HD - 1);
        dst[(((size_t)(b_ * HEADS + h) * SEQ + s) * HD) + d] = (_Float16)y;
      }
    }
  }
}

// =============== Kernel 2: flash attention (f16 in head layout -> f16 ctx [B*S, D]) ===============
__global__ void __launch_bounds__(128) flash_attn_kernel(
    const _Float16* __restrict__ Qh, const _Float16* __restrict__ Kh,
    const _Float16* __restrict__ Vh, _Float16* __restrict__ ctx) {
  __shared__ __align__(16) _Float16 Kt[64 * LDP];       // [k][d] row-major
  __shared__ __align__(16) _Float16 VtT[64 * LDP];      // [d][k] transposed
  __shared__ __align__(16) _Float16 Pt[4][16 * LDP];

  int tid = threadIdx.x, lane = tid & 31, wv = tid >> 5;
  int blk = blockIdx.x;
  int qt = blk & 31;          // 32 q-tiles of 64 rows
  int bh = blk >> 5;          // b*HEADS + h
  int b_ = bh >> 4, h = bh & 15;

  const _Float16* Qb = Qh + (size_t)bh * SEQ * HD;
  const _Float16* Kb = Kh + (size_t)bh * SEQ * HD;
  const _Float16* Vb = Vh + (size_t)bh * SEQ * HD;

  int q0 = qt * 64 + wv * 16;
  v16h qa0 = load_a_f16g(Qb, HD, q0, 0, lane);     // Q rows, d 0..31
  v16h qa1 = load_a_f16g(Qb, HD, q0, 32, lane);    // Q rows, d 32..63

  v8f O[4] = {{0}, {0}, {0}, {0}};
  float mrow[8], lrow[8];
#pragma unroll
  for (int r = 0; r < 8; ++r) { mrow[r] = -1e30f; lrow[r] = 0.0f; }

  for (int kt = 0; kt < SEQ / 64; ++kt) {
    __syncthreads();
    // ---- stage K (row-major, 16B copies) and V (transposed) into LDS ----
    const uint4* Kg4 = (const uint4*)(Kb + (size_t)kt * 64 * HD);
    const uint4* Vg4 = (const uint4*)(Vb + (size_t)kt * 64 * HD);
    uint4* KtQ = (uint4*)Kt;
#pragma unroll
    for (int it = 0; it < 4; ++it) {
      int idx = tid + it * 128;               // 0..511 (tile = 512 uint4)
      int r = idx >> 3, cc = idx & 7;         // k row, uint4-within-row
      KtQ[r * 9 + cc] = Kg4[idx];             // LDS row = 9 uint4 (144B)
      uint4 raw = Vg4[idx];                   // 8 halves of V[k=r][d0..d0+7]
      const _Float16* hv = (const _Float16*)&raw;
      int d0 = cc * 8;
#pragma unroll
      for (int j = 0; j < 8; ++j) VtT[(d0 + j) * LDP + r] = hv[j];
    }
    if (kt + 1 < SEQ / 64) {                  // hint next tiles toward the caches
      __builtin_prefetch((const char*)(Kg4 + 512) + tid * 64, 0, 0);
      __builtin_prefetch((const char*)(Vg4 + 512) + tid * 64, 0, 0);
    }
    __syncthreads();

    // ---- scores: S(16x64) = Q(16x64) @ K^T, 8 WMMAs ----
    v8f Sf[4];
#pragma unroll
    for (int nf = 0; nf < 4; ++nf) {
      v16h bk0 = load_b_kt(Kt, nf * 16, 0, lane);
      v16h bk1 = load_b_kt(Kt, nf * 16, 32, lane);
      v8f s = {0};
      s = wmma16(qa0, bk0, s);
      s = wmma16(qa1, bk1, s);
      Sf[nf] = s;
    }

    // ---- online softmax (rows live in C-layout: 8 rows per lane, 16 lanes share a row) ----
#pragma unroll
    for (int r = 0; r < 8; ++r) {
      float mx = fmaxf(fmaxf(Sf[0][r], Sf[1][r]), fmaxf(Sf[2][r], Sf[3][r]));
#pragma unroll
      for (int off = 1; off < 16; off <<= 1) mx = fmaxf(mx, __shfl_xor(mx, off, 32));
      float mnew = fmaxf(mrow[r], mx);
      float corr = __expf(mrow[r] - mnew);
      float ls = 0.0f;
#pragma unroll
      for (int nf = 0; nf < 4; ++nf) {
        float p = __expf(Sf[nf][r] - mnew);
        Sf[nf][r] = p;
        ls += p;
      }
#pragma unroll
      for (int off = 1; off < 16; off <<= 1) ls += __shfl_xor(ls, off, 32);
      lrow[r] = lrow[r] * corr + ls;
      mrow[r] = mnew;
#pragma unroll
      for (int nf = 0; nf < 4; ++nf) O[nf][r] *= corr;
    }

    // ---- restage P through LDS: C-layout -> A-layout ----
    _Float16* Pw = &Pt[wv][0];
    int prow = (lane >> 4) << 3;
    int pcol = lane & 15;
#pragma unroll
    for (int r = 0; r < 8; ++r)
#pragma unroll
      for (int nf = 0; nf < 4; ++nf)
        Pw[(prow + r) * LDP + nf * 16 + pcol] = (_Float16)Sf[nf][r];
    __syncthreads();

    // ---- O += P(16x64) @ V(64x64), 8 WMMAs (V fragments now contiguous in LDS) ----
#pragma unroll
    for (int kf = 0; kf < 2; ++kf) {
      v16h pa = load_a_lds(Pw, LDP, kf * 32, lane);
#pragma unroll
      for (int nf = 0; nf < 4; ++nf) {
        v16h bv = load_b_vtT(VtT, kf * 32, nf * 16, lane);
        O[nf] = wmma16(pa, bv, O[nf]);
      }
    }
  }

  // ---- normalize and write ctx back in [B*S, HIDDEN] f16 (heads re-interleaved) ----
#pragma unroll
  for (int r = 0; r < 8; ++r) {
    float inv = 1.0f / lrow[r];
    int q = q0 + ((lane >> 4) << 3) + r;
#pragma unroll
    for (int nf = 0; nf < 4; ++nf) {
      int col = (h << 6) + nf * 16 + (lane & 15);
      ctx[(size_t)(b_ * SEQ + q) * HIDDEN + col] = (_Float16)(O[nf][r] * inv);
    }
  }
}

// =============== Kernel 3: output projection (f16 ctx x f16 Wo -> f32 out) ===============
__global__ void __launch_bounds__(128) out_proj_kernel(
    const _Float16* __restrict__ ctx, const _Float16* __restrict__ Woh,
    const float* __restrict__ bo, float* __restrict__ out) {
  int id = blockIdx.x;
  int nt = id & 15;
  int mt = id >> 4;
  int lane = threadIdx.x & 31, wv = threadIdx.x >> 5;
  int m0 = mt * 128 + wv * 32;          // 32 rows per wave
  int n0 = nt * 64;
  v8f acc[8] = {{0},{0},{0},{0},{0},{0},{0},{0}};
#pragma unroll 4
  for (int kb = 0; kb < HIDDEN; kb += 32) {
    v16h a0 = load_a_f16g(ctx, HIDDEN, m0, kb, lane);
    v16h a1 = load_a_f16g(ctx, HIDDEN, m0 + 16, kb, lane);
    v16h bfr[4];
#pragma unroll
    for (int nf = 0; nf < 4; ++nf) bfr[nf] = load_bT_f16(Woh, HIDDEN, n0 + nf * 16, kb, lane);
#pragma unroll
    for (int nf = 0; nf < 4; ++nf) acc[nf] = wmma16(a0, bfr[nf], acc[nf]);
#pragma unroll
    for (int nf = 0; nf < 4; ++nf) acc[4 + nf] = wmma16(a1, bfr[nf], acc[4 + nf]);
  }
  int rbase = (lane >> 4) << 3;
  int cl = lane & 15;
#pragma unroll
  for (int g = 0; g < 2; ++g) {
#pragma unroll
    for (int nf = 0; nf < 4; ++nf) {
#pragma unroll
      for (int r = 0; r < 8; ++r) {
        int m = m0 + g * 16 + rbase + r;
        int n = n0 + nf * 16 + cl;
        out[(size_t)m * HIDDEN + n] = acc[g * 4 + nf][r] + bo[n];
      }
    }
  }
}

extern "C" void kernel_launch(void* const* d_in, const int* in_sizes, int n_in,
                              void* d_out, int out_size, void* d_ws, size_t ws_size,
                              hipStream_t stream) {
  (void)in_sizes; (void)n_in; (void)out_size; (void)ws_size;
  const float* query = (const float*)d_in[0];
  const float* key   = (const float*)d_in[1];
  const float* value = (const float*)d_in[2];
  const float* Wq = (const float*)d_in[3];
  const float* bq = (const float*)d_in[4];
  const float* Wk = (const float*)d_in[5];
  const float* bk = (const float*)d_in[6];
  const float* Wv = (const float*)d_in[7];
  const float* bv = (const float*)d_in[8];
  const float* Wo = (const float*)d_in[9];
  const float* bo = (const float*)d_in[10];
  float* out = (float*)d_out;

  // ---- workspace layout (all f16, fully overwritten every call; 64 MB total) ----
  const size_t XE = (size_t)MROWS * HIDDEN;                 // 4,194,304 (activations)
  const size_t WE = (size_t)HIDDEN * HIDDEN;                // 1,048,576 (weights)
  _Float16* p   = (_Float16*)d_ws;
  _Float16* Xq  = p; p += XE;
  _Float16* Xk  = p; p += XE;
  _Float16* Xv  = p; p += XE;
  _Float16* Wqh = p; p += WE;
  _Float16* Wkh = p; p += WE;
  _Float16* Wvh = p; p += WE;
  _Float16* Woh = p; p += WE;
  _Float16* Qh  = p; p += XE;   // [B,H,S,Hd]
  _Float16* Kh  = p; p += XE;
  _Float16* Vh  = p; p += XE;
  _Float16* ctx = p; p += XE;   // [B*S, HIDDEN]

  // ---- stage 0: one-shot f32 -> f16 conversion of every GEMM operand ----
  const int X4 = (int)(XE / 4), W4 = (int)(WE / 4);
  cvt_f16_kernel<<<X4 / 256, 256, 0, stream>>>((const float4*)query, (v4h*)Xq, X4);
  cvt_f16_kernel<<<X4 / 256, 256, 0, stream>>>((const float4*)key,   (v4h*)Xk, X4);
  cvt_f16_kernel<<<X4 / 256, 256, 0, stream>>>((const float4*)value, (v4h*)Xv, X4);
  cvt_f16_kernel<<<W4 / 256, 256, 0, stream>>>((const float4*)Wq, (v4h*)Wqh, W4);
  cvt_f16_kernel<<<W4 / 256, 256, 0, stream>>>((const float4*)Wk, (v4h*)Wkh, W4);
  cvt_f16_kernel<<<W4 / 256, 256, 0, stream>>>((const float4*)Wv, (v4h*)Wvh, W4);
  cvt_f16_kernel<<<W4 / 256, 256, 0, stream>>>((const float4*)Wo, (v4h*)Woh, W4);

  // ---- stage 1: QKV projections (3 matrices x 32 m-tiles x 16 n-tiles) ----
  qkv_proj_kernel<<<3 * 32 * 16, 128, 0, stream>>>(Xq, Xk, Xv, Wqh, Wkh, Wvh,
                                                   bq, bk, bv, Qh, Kh, Vh);
  // ---- stage 2: attention (B*H*(S/64) blocks) ----
  flash_attn_kernel<<<BATCH * HEADS * (SEQ / 64), 128, 0, stream>>>(Qh, Kh, Vh, ctx);
  // ---- stage 3: output projection ----
  out_proj_kernel<<<32 * 16, 128, 0, stream>>>(ctx, Woh, bo, out);
}
